// Decoder_16492674417365
// MI455X (gfx1250) — matrix operand order repaired and checked by reading, not compiled, not run on hardware
//
#include <hip/hip_runtime.h>
#include <hip/hip_bf16.h>

// ---------------------------------------------------------------------------
// LSTM decoder for MI455X (gfx1250, wave32, WMMA, async LDS staging).
//   gates[B,4H] = [c|h][B,2K] @ Wcat^T[2K,4H] + bias   (bf16 WMMA, f32 acc)
//   pointwise LSTM update (fp32 state) + repack next-step A fragments (bf16)
//   out[B,t,O]  = h2[B,H] @ W_out^T[H,O] + b_out       (bf16 WMMA, f32 acc)
// Weights are converted once into WMMA-fragment-swizzled bf16 buffers; they
// stay L2-resident (25 MB << 192 MB). Gate GEMM stages fragments into LDS
// (double buffered, async-to-LDS) for ~43 FLOP/byte from L2.
// ---------------------------------------------------------------------------

typedef __bf16 v16bf  __attribute__((ext_vector_type(16)));
typedef float  v8f    __attribute__((ext_vector_type(8)));
typedef __bf16 bf16x2 __attribute__((ext_vector_type(2)));

// async-to-LDS builtin operand types: b128 payload = int __vector(4),
// global side in AS1, LDS side in AS3 (per hipcc diagnostic of the builtin).
typedef int v4i __attribute__((vector_size(16)));
typedef __attribute__((address_space(1))) v4i* v4i_gp;
typedef __attribute__((address_space(3))) v4i* v4i_lp;

#define BB 256    // batch (M of gate GEMM)
#define HH 1024   // hidden
#define GG 4096   // 4*H  (N of gate GEMM)
#define KK 2048   // [c|h] concat (K of gate GEMM)
#define OO 128    // output dim
#define TT 512    // seq len
#define START_VAL 128.0f

#if __has_builtin(__builtin_amdgcn_global_load_async_to_lds_b128)
#define HAVE_ASYNC_LDS 1
#else
#define HAVE_ASYNC_LDS 0
#endif

__device__ __forceinline__ void wait_async_all() {
#if HAVE_ASYNC_LDS
#if __has_builtin(__builtin_amdgcn_s_wait_asynccnt)
  __builtin_amdgcn_s_wait_asynccnt(0);
#else
  asm volatile("s_wait_asynccnt 0x0" ::: "memory");
#endif
#endif
}

__device__ __forceinline__ void copy16_to_lds(const char* g, char* l) {
#if HAVE_ASYNC_LDS
  __builtin_amdgcn_global_load_async_to_lds_b128(
      (v4i_gp)(const void*)g, (v4i_lp)(void*)l, 0, 0);
#else
  *(uint4*)l = *(const uint4*)g;
#endif
}

// --- WMMA 16-bit fragment element indexing (CDNA5 ISA 7.12.2) ----------
// A (16x32, MxK): lanes 0-15 hold K in {0..7,16..23}, lanes 16-31 hold
// K in {8..15,24..31}; each VGPR holds 2 packed bf16 (K even/odd).
__device__ __forceinline__ int a_elem(int mloc, int kloc) {
  int grp  = (kloc >> 3) & 1;
  int lane = mloc + (grp << 4);
  int vgpr = ((kloc & 7) >> 1) + ((kloc >> 4) << 2);
  return (lane << 4) + (vgpr << 1) + (kloc & 1);
}
// B (32x16, KxN): lanes 0-15 hold K in {0..7,16..23} (col = lane), lanes 16-31 K in {8..15,24..31}.
__device__ __forceinline__ int b_elem(int kloc, int nloc) {
  int lane = nloc + (((kloc >> 3) & 1) << 4);
  int vgpr = ((kloc & 7) >> 1) + ((kloc >> 4) << 2);
  return (lane << 4) + (vgpr << 1) + (kloc & 1);
}

__device__ __forceinline__ float sigf(float x) {
  return 1.0f / (1.0f + __expf(-x));
}

__device__ __forceinline__ v8f wmma_bf16(v16bf a, v16bf b, v8f c) {
  return __builtin_amdgcn_wmma_f32_16x16x32_bf16(false, a, false, b,
                                                 (short)0, c, false, false);
}

// --- one-time weight packing ------------------------------------------------
__global__ void pack_gate_w(const float* __restrict__ Wih,
                            const float* __restrict__ Whh,
                            __bf16* __restrict__ Bp) {
  int idx = blockIdx.x * 256 + threadIdx.x;
  if (idx >= GG * KK) return;
  int n = idx / KK, k = idx - n * KK;
  float v = (k < HH) ? Wih[n * HH + k] : Whh[n * HH + (k - HH)];
  int nt = n >> 4, kt = k >> 5;
  Bp[(((nt << 6) + kt) << 9) + b_elem(k & 31, n & 15)] = (__bf16)v;
}

__global__ void pack_out_w(const float* __restrict__ Wout,
                           __bf16* __restrict__ Wp) {
  int idx = blockIdx.x * 256 + threadIdx.x;
  if (idx >= OO * HH) return;
  int n = idx / HH, k = idx - n * HH;
  int nt = n >> 4, kt = k >> 5;  // 32 K-tiles
  Wp[(((nt << 5) + kt) << 9) + b_elem(k & 31, n & 15)] = (__bf16)Wout[idx];
}

__global__ void init_state(const float* __restrict__ bih,
                           const float* __restrict__ bhh,
                           float* __restrict__ bias,
                           __bf16* __restrict__ Ap,
                           float* __restrict__ cbuf) {
  int idx = blockIdx.x * 256 + threadIdx.x;
  if (idx < GG) bias[idx] = bih[idx] + bhh[idx];
  if (idx < BB * KK) {
    int m = idx / KK, k = idx - m * KK;
    float v = (k < HH) ? START_VAL : 0.0f;
    int mt = m >> 4, kt = k >> 5;
    Ap[(((mt << 6) + kt) << 9) + a_elem(m & 15, k & 31)] = (__bf16)v;
  }
  if (idx < BB * HH) cbuf[idx] = 0.0f;
}

// --- gate GEMM: [256,2048]x[2048,4096] + bias -------------------------------
// 8 wave32s per block (2M x 4N), wave tile 32x32 (4 accumulators).
// Block tile 64M x 128N; grid (4, 32). LDS stage per K-step: 4 A-tiles +
// 8 B-tiles = 12 KB, double buffered (24 KB of 320 KB/WGP).
#define STAGE_BYTES (12 * 1024)

__device__ __forceinline__ void stage_frags(const char* Ab, const char* Bb,
                                            int mtB, int ntB, int kt,
                                            char* dst, int tid) {
#pragma unroll
  for (int i = 0; i < 3; ++i) {
    int c    = tid + (i << 8);     // 0..767 chunks of 16B
    int tile = c >> 6;             // 0..11
    int off  = (c & 63) << 4;      // byte offset within 1KB tile
    const char* src = (tile < 4)
        ? Ab + ((((mtB + tile) << 6) + kt) << 10) + off
        : Bb + ((((ntB + tile - 4) << 6) + kt) << 10) + off;
    if (i == 0)  // pull kt+4 toward near cache; speculative, silently dropped
      __builtin_prefetch((const void*)(src + 4096), 0, 3);
    copy16_to_lds(src, dst + (c << 4));
  }
}

__global__ void __launch_bounds__(256)
gemm_gates(const __bf16* __restrict__ Ap, const __bf16* __restrict__ Bp,
           const float* __restrict__ bias, float* __restrict__ gates) {
  __shared__ __attribute__((aligned(32))) char lds[2][STAGE_BYTES];

  const int tid  = threadIdx.x;
  const int lane = tid & 31;
  const int w    = tid >> 5;
  const int wm   = w & 1;        // 2 m-tile pairs
  const int wn   = w >> 1;       // 4 n-tile pairs
  const int mtB  = blockIdx.x << 2;  // 4 m-tiles per block
  const int ntB  = blockIdx.y << 3;  // 8 n-tiles per block

  const char* Ab = (const char*)Ap;
  const char* Bb = (const char*)Bp;

  const int nloc = lane & 15;
  const float bv0 = bias[((ntB + (wn << 1) + 0) << 4) + nloc];
  const float bv1 = bias[((ntB + (wn << 1) + 1) << 4) + nloc];
  v8f acc00, acc01, acc10, acc11;
#pragma unroll
  for (int r = 0; r < 8; ++r) {
    acc00[r] = bv0; acc01[r] = bv1; acc10[r] = bv0; acc11[r] = bv1;
  }

  const int aOff0 = ((wm << 1) + 0) * 1024 + (lane << 5);
  const int aOff1 = ((wm << 1) + 1) * 1024 + (lane << 5);
  const int bOff0 = (4 + (wn << 1) + 0) * 1024 + (lane << 5);
  const int bOff1 = (4 + (wn << 1) + 1) * 1024 + (lane << 5);

  stage_frags(Ab, Bb, mtB, ntB, 0, lds[0], tid);
  wait_async_all();
  __syncthreads();

  for (int kt = 0; kt < 64; ++kt) {
    char* cur = lds[kt & 1];
    if (kt + 1 < 64)
      stage_frags(Ab, Bb, mtB, ntB, kt + 1, lds[(kt + 1) & 1], tid);

    v16bf a0 = *(const v16bf*)(cur + aOff0);
    v16bf a1 = *(const v16bf*)(cur + aOff1);
    v16bf b0 = *(const v16bf*)(cur + bOff0);
    v16bf b1 = *(const v16bf*)(cur + bOff1);
    acc00 = wmma_bf16(a0, b0, acc00);
    acc01 = wmma_bf16(a0, b1, acc01);
    acc10 = wmma_bf16(a1, b0, acc10);
    acc11 = wmma_bf16(a1, b1, acc11);

    wait_async_all();
    __syncthreads();
  }

  // C/D layout: m = r + 8*(lane>>4), n = lane&15
  const int mofs = (lane >> 4) << 3;
  const int mt0  = mtB + (wm << 1);
  const int nt0  = ntB + (wn << 1);
#pragma unroll
  for (int r = 0; r < 8; ++r) {
    int row0 = ((mt0 + 0) << 4) + mofs + r;
    int row1 = ((mt0 + 1) << 4) + mofs + r;
    gates[row0 * GG + ((nt0 + 0) << 4) + nloc] = acc00[r];
    gates[row0 * GG + ((nt0 + 1) << 4) + nloc] = acc01[r];
    gates[row1 * GG + ((nt0 + 0) << 4) + nloc] = acc10[r];
    gates[row1 * GG + ((nt0 + 1) << 4) + nloc] = acc11[r];
  }
}

// --- pointwise LSTM update + fragment repack --------------------------------
__global__ void __launch_bounds__(256)
lstm_pointwise(const float* __restrict__ gates, float* __restrict__ cbuf,
               __bf16* __restrict__ Ap, __bf16* __restrict__ Aout) {
  int idx = blockIdx.x * 256 + threadIdx.x;   // 256*512 threads
  int b = idx >> 9;
  int h = (idx & 511) << 1;

  const float2 ig = *(const float2*)&gates[b * GG + h];
  const float2 fg = *(const float2*)&gates[b * GG + HH + h];
  const float2 gg = *(const float2*)&gates[b * GG + 2 * HH + h];
  const float2 og = *(const float2*)&gates[b * GG + 3 * HH + h];
  float2 c = *(float2*)&cbuf[b * HH + h];

  float c20 = sigf(fg.x) * c.x + sigf(ig.x) * tanhf(gg.x);
  float c21 = sigf(fg.y) * c.y + sigf(ig.y) * tanhf(gg.y);
  float h20 = sigf(og.x) * tanhf(c20);
  float h21 = sigf(og.y) * tanhf(c21);
  *(float2*)&cbuf[b * HH + h] = make_float2(c20, c21);

  const int mt = b >> 4, ml = b & 15;
  {  // A first half: k = h  (next input = c2)
    int kt = h >> 5, kl = h & 31;
    bf16x2 v; v.x = (__bf16)c20; v.y = (__bf16)c21;
    *(bf16x2*)&Ap[(((mt << 6) + kt) << 9) + a_elem(ml, kl)] = v;
  }
  {  // A second half: k = HH + h (next hidden = h2)
    int k = HH + h, kt = k >> 5, kl = k & 31;
    bf16x2 v; v.x = (__bf16)h20; v.y = (__bf16)h21;
    *(bf16x2*)&Ap[(((mt << 6) + kt) << 9) + a_elem(ml, kl)] = v;
  }
  {  // A_out: h2, K-space of 1024 (32 K-tiles)
    int kt = h >> 5, kl = h & 31;
    bf16x2 v; v.x = (__bf16)h20; v.y = (__bf16)h21;
    *(bf16x2*)&Aout[(((mt << 5) + kt) << 9) + a_elem(ml, kl)] = v;
  }
}

// --- output GEMM: [256,1024]x[1024,128] + b_out -----------------------------
__global__ void __launch_bounds__(256)
gemm_out(const __bf16* __restrict__ Aout, const __bf16* __restrict__ Wp,
         const float* __restrict__ bout, float* __restrict__ out, int t) {
  const int lane = threadIdx.x & 31;
  const int nt   = threadIdx.x >> 5;  // 0..7
  const int mt   = blockIdx.x;        // 0..15
  const v16bf* A  = (const v16bf*)Aout;
  const v16bf* Bm = (const v16bf*)Wp;

  const float bv = bout[(nt << 4) + (lane & 15)];
  v8f acc;
#pragma unroll
  for (int r = 0; r < 8; ++r) acc[r] = bv;

#pragma unroll 4
  for (int kt = 0; kt < 32; ++kt) {
    v16bf a = A[(((mt << 5) + kt) << 5) + lane];
    v16bf b = Bm[(((nt << 5) + kt) << 5) + lane];
    acc = wmma_bf16(a, b, acc);
  }

  const int nloc = lane & 15;
  const int mofs = (lane >> 4) << 3;
#pragma unroll
  for (int r = 0; r < 8; ++r) {
    int row = (mt << 4) + mofs + r;  // batch index
    out[(size_t)row * (TT * OO) + t * OO + (nt << 4) + nloc] = acc[r];
  }
}

// ---------------------------------------------------------------------------
extern "C" void kernel_launch(void* const* d_in, const int* in_sizes, int n_in,
                              void* d_out, int out_size, void* d_ws, size_t ws_size,
                              hipStream_t stream) {
  (void)in_sizes; (void)n_in; (void)out_size; (void)ws_size;
  // inputs: x(unused), W_ih, W_hh, b_ih, b_hh, W_out, b_out  (all fp32)
  const float* W_ih = (const float*)d_in[1];
  const float* W_hh = (const float*)d_in[2];
  const float* b_ih = (const float*)d_in[3];
  const float* b_hh = (const float*)d_in[4];
  const float* Wout = (const float*)d_in[5];
  const float* bout = (const float*)d_in[6];
  float* out = (float*)d_out;

  char* p = (char*)d_ws;
  __bf16* Bpack = (__bf16*)p; p += (size_t)GG * KK * 2;   // 16.8 MB
  __bf16* WoutP = (__bf16*)p; p += (size_t)OO * HH * 2;   // 256 KB
  __bf16* Apack = (__bf16*)p; p += (size_t)BB * KK * 2;   // 1 MB
  __bf16* AoutP = (__bf16*)p; p += (size_t)BB * HH * 2;   // 512 KB
  float*  bias  = (float*)p;  p += (size_t)GG * 4;        // 16 KB
  float*  cbuf  = (float*)p;  p += (size_t)BB * HH * 4;   // 1 MB
  float*  gates = (float*)p;  p += (size_t)BB * GG * 4;   // 4 MB

  pack_gate_w<<<(GG * KK + 255) / 256, 256, 0, stream>>>(W_ih, W_hh, Bpack);
  pack_out_w<<<(OO * HH + 255) / 256, 256, 0, stream>>>(Wout, WoutP);
  init_state<<<(BB * KK + 255) / 256, 256, 0, stream>>>(b_ih, b_hh, bias,
                                                        Apack, cbuf);

  dim3 ggrid(4, 32);
  // priming step: input = START_VAL constant, h0 = c0 = 0
  gemm_gates<<<ggrid, 256, 0, stream>>>(Apack, Bpack, bias, gates);
  lstm_pointwise<<<512, 256, 0, stream>>>(gates, cbuf, Apack, AoutP);

  for (int t = 0; t < TT; ++t) {
    gemm_gates<<<ggrid, 256, 0, stream>>>(Apack, Bpack, bias, gates);
    lstm_pointwise<<<512, 256, 0, stream>>>(gates, cbuf, Apack, AoutP);
    gemm_out<<<16, 256, 0, stream>>>(AoutP, WoutP, bout, out, t);
  }
}
